// CausalTransition_60421599920464
// MI455X (gfx1250) — compile-verified, matrix operand order checked by reading, not run on hardware
//
#include <hip/hip_runtime.h>
#include <hip/hip_bf16.h>

#define BB 8
#define NN 64
#define DD 256
#define AA 16
#define LL 800
#define NP 80   // 65 graph nodes padded to 80 (multiple of 16)

typedef _Float16 half_t;
typedef __attribute__((ext_vector_type(16))) _Float16 v16h;
typedef __attribute__((ext_vector_type(8)))  _Float16 v8h;
typedef __attribute__((ext_vector_type(8)))  float    v8f;

// ---------------------------------------------------------------------------
// WMMA GEMM: C[M,N] = A[M,K](f16) @ Bt[N,K]^T(f16) + bias[N], f32 out.
// One wave per (16*MT)x16 tile; B fragment reused across MT M-subtiles.
// K is a compile-time constant so all per-subtile addresses fold into 24-bit
// instruction offsets. Explicit double-buffering: k-step k+32 fragments are
// loaded into a fresh register set while WMMAs consume step k, keeping both
// sets live so the scheduler can overlap loads with matrix ops (partial
// s_wait_loadcnt instead of a full drain per WMMA).
// ---------------------------------------------------------------------------
template <int MT, int K>
__global__ void __launch_bounds__(32) k_gemm(
    const half_t* __restrict__ A, const half_t* __restrict__ Bt,
    const float* __restrict__ bias, float* __restrict__ C,
    int M, int N,
    long aBatch, long bBatch, long biasBatch, long cBatch,
    const int* __restrict__ bSel)
{
    const int z  = blockIdx.z;
    const int bz = bSel ? bSel[z] : z;
    A  += (long)z  * aBatch;
    Bt += (long)bz * bBatch;
    C  += (long)z  * cBatch;
    const float* biasp = bias ? (bias + (long)bz * biasBatch) : nullptr;

    const int mt   = blockIdx.y * (16 * MT);
    const int nt   = blockIdx.x << 4;
    const int lane = threadIdx.x & 31;
    const int r16  = lane & 15;
    const int hi   = lane >> 4;

    const half_t* arow = A  + (long)(mt + r16) * K + 8 * hi;
    const half_t* brow = Bt + (long)(nt + r16) * K + 8 * hi;

    v8f acc[MT];
    #pragma unroll
    for (int m = 0; m < MT; ++m) acc[m] = (v8f){};

    // ---- prologue: load K-step 0 ----
    v8h b0c = *(const v8h*)(brow);
    v8h b1c = *(const v8h*)(brow + 16);
    v8h a0c[MT], a1c[MT];
    #pragma unroll
    for (int m = 0; m < MT; ++m) {
        a0c[m] = *(const v8h*)(arow + m * 16 * K);
        a1c[m] = *(const v8h*)(arow + m * 16 * K + 16);
    }

    for (int k0 = 32; k0 < K; k0 += 32) {
        // prefetch the weight stream ahead (global_prefetch_b8; speculative)
        __builtin_prefetch((const void*)(brow + k0 + 224), 0, 1);

        // ---- issue next K-step loads into a fresh register set ----
        v8h b0n = *(const v8h*)(brow + k0);
        v8h b1n = *(const v8h*)(brow + k0 + 16);
        v8h a0n[MT], a1n[MT];
        #pragma unroll
        for (int m = 0; m < MT; ++m) {
            a0n[m] = *(const v8h*)(arow + m * 16 * K + k0);
            a1n[m] = *(const v8h*)(arow + m * 16 * K + k0 + 16);
        }

        // ---- WMMAs on the current set ----
        v16h b;
        #pragma unroll
        for (int i = 0; i < 8; ++i) { b[i] = b0c[i]; b[i + 8] = b1c[i]; }
        #pragma unroll
        for (int m = 0; m < MT; ++m) {
            v16h a;
            #pragma unroll
            for (int i = 0; i < 8; ++i) { a[i] = a0c[m][i]; a[i + 8] = a1c[m][i]; }
            acc[m] = __builtin_amdgcn_wmma_f32_16x16x32_f16(
                false, a, false, b, (short)0, acc[m], false, false);
        }

        // ---- rotate buffers ----
        b0c = b0n; b1c = b1n;
        #pragma unroll
        for (int m = 0; m < MT; ++m) { a0c[m] = a0n[m]; a1c[m] = a1n[m]; }
    }

    // ---- epilogue: last K-step ----
    {
        v16h b;
        #pragma unroll
        for (int i = 0; i < 8; ++i) { b[i] = b0c[i]; b[i + 8] = b1c[i]; }
        #pragma unroll
        for (int m = 0; m < MT; ++m) {
            v16h a;
            #pragma unroll
            for (int i = 0; i < 8; ++i) { a[i] = a0c[m][i]; a[i + 8] = a1c[m][i]; }
            acc[m] = __builtin_amdgcn_wmma_f32_16x16x32_f16(
                false, a, false, b, (short)0, acc[m], false, false);
        }
    }

    #pragma unroll
    for (int m = 0; m < MT; ++m) {
        #pragma unroll
        for (int r = 0; r < 8; ++r) {
            int row = mt + m * 16 + r + 8 * hi;
            int col = nt + r16;
            if (row < M && col < N) {
                float v = acc[m][r];
                if (biasp) v += biasp[col];
                C[(long)row * N + col] = v;
            }
        }
    }
}

// src [z][K][N] f32 (row-major)  ->  dst [z][N][K] f16
__global__ void k_transpose(const float* __restrict__ src, half_t* __restrict__ dst,
                            int K, int N, long sBatch, long dBatch)
{
    int z = blockIdx.z;
    src += (long)z * sBatch; dst += (long)z * dBatch;
    int n = blockIdx.x * 16 + threadIdx.x;
    int k = blockIdx.y * 16 + threadIdx.y;
    if (n < N && k < K) dst[(long)n * K + k] = (half_t)src[(long)k * N + n];
}

__global__ void k_argmax(const float* __restrict__ action, int* __restrict__ idx)
{
    int b = threadIdx.x;
    if (b >= BB) return;
    float best = action[b * AA]; int bi = 0;
    for (int a = 1; a < AA; ++a) {
        float v = action[b * AA + a];
        if (v > best) { best = v; bi = a; }
    }
    idx[b] = bi;
}

// mask[b,n] = hard gumbel sample of Bernoulli(iml) (index-1 one-hot value)
__global__ void k_mask(const float* __restrict__ latent, const float* __restrict__ action,
                       const float* __restrict__ Wm, const float* __restrict__ bm,
                       const float* __restrict__ g_mask, float* __restrict__ mask)
{
    int t = blockIdx.x * 256 + threadIdx.x;      // t = b*NN + n
    if (t >= BB * NN) return;
    int b = t / NN;
    float iml = 0.f;
    for (int d = 0; d < DD; ++d) {
        float im = bm[d];
        for (int a = 0; a < AA; ++a) im += action[b * AA + a] * Wm[a * DD + d];
        im = 1.f / (1.f + __expf(-im));
        iml += latent[(long)t * DD + d] * im;
    }
    float l0 = __logf(fmaxf(1.f - iml, 1e-4f));
    float l1 = __logf(fmaxf(iml,       1e-4f));
    mask[t] = (l1 + g_mask[t * 2 + 1] > l0 + g_mask[t * 2 + 0]) ? 1.f : 0.f;
}

__global__ void k_plh(const float* __restrict__ latent, const float* __restrict__ pe,
                      half_t* __restrict__ plh)
{
    int t = blockIdx.x * 256 + threadIdx.x;
    if (t >= BB * NN * DD) return;
    int d = t % DD; int n = (t / DD) % NN;
    plh[t] = (half_t)(latent[t] + pe[n * DD + d]);
}

__global__ void k_nodes(const float* __restrict__ latent, const float* __restrict__ action,
                        const float* __restrict__ Wad, const float* __restrict__ bad,
                        half_t* __restrict__ nodesh)
{
    int t = blockIdx.x * 256 + threadIdx.x;
    if (t >= BB * NP * DD) return;
    int d = t % DD; int r = (t / DD) % NP; int b = t / (DD * NP);
    float v;
    if (r < NN)       v = latent[((long)b * NN + r) * DD + d];
    else if (r == NN) {
        v = bad[d];
        for (int a = 0; a < AA; ++a) v += action[b * AA + a] * Wad[a * DD + d];
    } else v = 0.f;
    nodesh[t] = (half_t)v;
}

// One wave per (b,i,j): adjacency via factored pair MLPs + gumbel-hard sample.
__global__ void k_pair(const float* __restrict__ U,  const float* __restrict__ V,
                       const float* __restrict__ Ue, const float* __restrict__ Ve,
                       const float* __restrict__ W02, const float* __restrict__ b02,
                       const float* __restrict__ We2, const float* __restrict__ be2,
                       const int* __restrict__ idx,  const float* __restrict__ mask,
                       const float* __restrict__ g_adj, float* __restrict__ adjacency)
{
    int wid  = (blockIdx.x * blockDim.x + threadIdx.x) >> 5;
    int lane = threadIdx.x & 31;
    if (wid >= BB * NN * NN) return;
    int j = wid % NN; int i = (wid / NN) % NN; int b = wid / (NN * NN);
    int e = idx[b];
    const float* u  = U  + ((long)b * NN + i) * LL;  // already has +b01
    const float* v  = V  + ((long)b * NN + j) * LL;
    const float* ue = Ue + ((long)b * NN + i) * LL;  // already has +be1[e]
    const float* ve = Ve + ((long)b * NN + j) * LL;
    const float* w2 = We2 + (long)e * LL;
    float s0 = 0.f, s1 = 0.f;
    for (int l = lane; l < LL; l += 32) {
        float h0 = u[l] + v[l];   h0 = h0 > 0.f ? h0 : 0.01f * h0;
        float h1 = ue[l] + ve[l]; h1 = h1 > 0.f ? h1 : 0.01f * h1;
        s0 += h0 * W02[l];
        s1 += h1 * w2[l];
    }
    for (int o = 16; o; o >>= 1) { s0 += __shfl_down(s0, o, 32); s1 += __shfl_down(s1, o, 32); }
    if (lane == 0) {
        float no_inter = 1.f / (1.f + __expf(-(s0 + b02[0])));
        float inter    = 1.f / (1.f + __expf(-(s1 + be2[e])));
        float mk = mask[b * NN + i];
        float p  = no_inter * (1.f - mk) + inter * mk;
        float l0 = __logf(fmaxf(1.f - p, 1e-4f));
        float l1 = __logf(fmaxf(p,       1e-4f));
        const float* g = g_adj + (long)wid * 2;
        adjacency[wid] = (l1 + g[1] > l0 + g[0]) ? 1.f : 0.f;
    }
}

// Global (all-batch) mean of nonzero padded-adjacency entries (self-loop fill).
__global__ void k_mean(const float* __restrict__ adjacency, float* __restrict__ meanOut)
{
    __shared__ float ss[256], sca[256];
    float sum = 0.f, cnt = 0.f;
    for (int t = threadIdx.x; t < BB * 65 * 65; t += 256) {
        int tt = t; int t2 = tt % 65; tt /= 65; int s = tt % 65; int b = tt / 65;
        float pnl;
        if (s == t2)       pnl = 0.f;
        else if (t2 == NN) pnl = (s < NN) ? 1.f : 0.f;
        else if (s == NN)  pnl = 0.f;
        else               pnl = adjacency[((long)b * NN + s) * NN + t2];
        sum += pnl;
        cnt += (pnl != 0.f) ? 1.f : 0.f;
    }
    ss[threadIdx.x] = sum; sca[threadIdx.x] = cnt;
    __syncthreads();
    for (int o = 128; o; o >>= 1) {
        if (threadIdx.x < o) { ss[threadIdx.x] += ss[threadIdx.x + o];
                               sca[threadIdx.x] += sca[threadIdx.x + o]; }
        __syncthreads();
    }
    if (threadIdx.x == 0) meanOut[0] = ss[0] / fmaxf(sca[0], 1.f);
}

// GATv2 attention + aggregation for one (b, target t); relu; f16 out (padded).
__global__ void __launch_bounds__(256) k_gatt(
    const float* __restrict__ gl, const float* __restrict__ gr,
    const float* __restrict__ adjacency, const float* __restrict__ meanPtr,
    const float* __restrict__ We, const float* __restrict__ att,
    const float* __restrict__ bvec, half_t* __restrict__ out)
{
    int t = blockIdx.x, b = blockIdx.y;
    gl  += (long)b * NP * LL;
    gr  += (long)b * NP * LL;
    out += ((long)b * NP + t) * LL;
    if (t >= 65) {                      // padding rows stay zero
        for (int c = threadIdx.x; c < LL; c += 256) out[c] = (half_t)0.f;
        return;
    }
    __shared__ float sc[65];
    __shared__ float alpha[65];
    const float mean = meanPtr[0];
    const int lane = threadIdx.x & 31, wv = threadIdx.x >> 5;
    const float* grt = gr + (long)t * LL;
    for (int s = wv; s < 65; s += 8) {
        float pnl;
        if (s == t)       pnl = 0.f;
        else if (t == NN) pnl = (s < NN) ? 1.f : 0.f;
        else if (s == NN) pnl = 0.f;
        else              pnl = adjacency[((long)b * NN + s) * NN + t];
        bool  msk = (pnl != 0.f) || (s == t);
        float aa  = (s == t) ? mean : pnl;
        const float* gls = gl + (long)s * LL;
        float sum = 0.f;
        for (int c = lane; c < LL; c += 32) {
            float z = gls[c] + grt[c] + aa * We[c];
            z = z > 0.f ? z : 0.2f * z;           // leaky_relu(0.2)
            sum += z * att[c];
        }
        for (int o = 16; o; o >>= 1) sum += __shfl_down(sum, o, 32);
        if (lane == 0) sc[s] = msk ? sum : -1e9f;
    }
    __syncthreads();
    if (threadIdx.x == 0) {                       // softmax over sources s
        float mx = -1e30f;
        for (int s = 0; s < 65; ++s) mx = fmaxf(mx, sc[s]);
        float sm = 0.f;
        for (int s = 0; s < 65; ++s) { float e = __expf(sc[s] - mx); alpha[s] = e; sm += e; }
        float inv = 1.f / sm;
        for (int s = 0; s < 65; ++s) alpha[s] *= inv;
    }
    __syncthreads();
    for (int c = threadIdx.x; c < LL; c += 256) {
        float acc = bvec[c];
        for (int s = 0; s < 65; ++s) acc += alpha[s] * gl[(long)s * LL + c];
        out[c] = (half_t)fmaxf(acc, 0.f);         // relu
    }
}

// Row softmax over D=256; drop action node (n < 64 only).
__global__ void __launch_bounds__(256) k_out(const float* __restrict__ logits,
                                             float* __restrict__ out)
{
    int n = blockIdx.x, b = blockIdx.y, d = threadIdx.x;
    const float* row = logits + ((long)b * NP + n) * DD;
    __shared__ float sh[256];
    float v = row[d];
    sh[d] = v; __syncthreads();
    for (int o = 128; o; o >>= 1) { if (d < o) sh[d] = fmaxf(sh[d], sh[d + o]); __syncthreads(); }
    float mx = sh[0]; __syncthreads();
    float e = __expf(v - mx);
    sh[d] = e; __syncthreads();
    for (int o = 128; o; o >>= 1) { if (d < o) sh[d] += sh[d + o]; __syncthreads(); }
    out[((long)b * NN + n) * DD + d] = e / sh[0];
}

// ---------------------------------------------------------------------------
extern "C" void kernel_launch(void* const* d_in, const int* in_sizes, int n_in,
                              void* d_out, int out_size, void* d_ws, size_t ws_size,
                              hipStream_t stream)
{
    const float* latent = (const float*)d_in[0];
    const float* action = (const float*)d_in[1];
    const float* pe     = (const float*)d_in[2];
    const float* g_mask = (const float*)d_in[3];
    const float* g_adj  = (const float*)d_in[4];
    const float* Wad    = (const float*)d_in[5];
    const float* bad    = (const float*)d_in[6];
    const float* Wm     = (const float*)d_in[7];
    const float* bm     = (const float*)d_in[8];
    const float* W01    = (const float*)d_in[9];
    const float* b01    = (const float*)d_in[10];
    const float* W02    = (const float*)d_in[11];
    const float* b02    = (const float*)d_in[12];
    const float* We1    = (const float*)d_in[13];
    const float* be1    = (const float*)d_in[14];
    const float* We2    = (const float*)d_in[15];
    const float* be2    = (const float*)d_in[16];
    const float* Wf     = (const float*)d_in[17];
    const float* bf     = (const float*)d_in[18];
    const float* gWl[3]  = {(const float*)d_in[19], (const float*)d_in[26], (const float*)d_in[33]};
    const float* gbl[3]  = {(const float*)d_in[20], (const float*)d_in[27], (const float*)d_in[34]};
    const float* gWr[3]  = {(const float*)d_in[21], (const float*)d_in[28], (const float*)d_in[35]};
    const float* gbr[3]  = {(const float*)d_in[22], (const float*)d_in[29], (const float*)d_in[36]};
    const float* gWe[3]  = {(const float*)d_in[23], (const float*)d_in[30], (const float*)d_in[37]};
    const float* gatt[3] = {(const float*)d_in[24], (const float*)d_in[31], (const float*)d_in[38]};
    const float* gb[3]   = {(const float*)d_in[25], (const float*)d_in[32], (const float*)d_in[39]};

    // ---- bump allocator over d_ws (256B aligned) ----
    char* wp = (char*)d_ws;
    auto balloc = [&](size_t bytes) -> void* {
        void* r = wp;
        wp += (bytes + 255) & ~(size_t)255;
        return r;
    };
    int*    idx    = (int*)   balloc(BB * sizeof(int));
    float*  mask   = (float*) balloc((size_t)BB * NN * 4);
    float*  meanv  = (float*) balloc(4);
    half_t* plh    = (half_t*)balloc((size_t)BB * NN * DD * 2);
    half_t* nodesh = (half_t*)balloc((size_t)BB * NP * DD * 2);
    float*  U      = (float*) balloc((size_t)BB * NN * LL * 4);
    float*  V      = (float*) balloc((size_t)BB * NN * LL * 4);
    float*  Ue     = (float*) balloc((size_t)BB * NN * LL * 4);
    float*  Ve     = (float*) balloc((size_t)BB * NN * LL * 4);
    float*  adj    = (float*) balloc((size_t)BB * NN * NN * 4);
    float*  glb    = (float*) balloc((size_t)BB * NP * LL * 4);
    float*  grb    = (float*) balloc((size_t)BB * NP * LL * 4);
    half_t* hA     = (half_t*)balloc((size_t)BB * NP * LL * 2);
    half_t* hB     = (half_t*)balloc((size_t)BB * NP * LL * 2);
    float*  logits = (float*) balloc((size_t)BB * NP * DD * 4);
    half_t* W01tT  = (half_t*)balloc((size_t)LL * DD * 2);
    half_t* W01tB  = (half_t*)balloc((size_t)LL * DD * 2);
    half_t* We1tT  = (half_t*)balloc((size_t)AA * LL * DD * 2);
    half_t* We1tB  = (half_t*)balloc((size_t)AA * LL * DD * 2);
    half_t* gWlt[3], *gWrt[3];
    for (int l = 0; l < 3; ++l) {
        size_t kk = (l == 0) ? DD : LL;
        gWlt[l] = (half_t*)balloc(LL * kk * 2);
        gWrt[l] = (half_t*)balloc(LL * kk * 2);
    }
    half_t* Wft = (half_t*)balloc((size_t)DD * LL * 2);

    // ---- prep ----
    k_argmax<<<1, 32, 0, stream>>>(action, idx);
    k_mask<<<(BB * NN + 255) / 256, 256, 0, stream>>>(latent, action, Wm, bm, g_mask, mask);
    k_plh<<<(BB * NN * DD + 255) / 256, 256, 0, stream>>>(latent, pe, plh);
    k_nodes<<<(BB * NP * DD + 255) / 256, 256, 0, stream>>>(latent, action, Wad, bad, nodesh);

    auto T = [&](const float* s, half_t* d, int K, int N, long sB, long dB, int z) {
        k_transpose<<<dim3((N + 15) / 16, (K + 15) / 16, z), dim3(16, 16), 0, stream>>>(s, d, K, N, sB, dB);
    };
    T(W01,            W01tT, DD, LL, 0, 0, 1);
    T(W01 + DD * LL,  W01tB, DD, LL, 0, 0, 1);
    T(We1,            We1tT, DD, LL, (long)2 * DD * LL, (long)LL * DD, AA);
    T(We1 + DD * LL,  We1tB, DD, LL, (long)2 * DD * LL, (long)LL * DD, AA);
    T(gWl[0], gWlt[0], DD, LL, 0, 0, 1);
    T(gWr[0], gWrt[0], DD, LL, 0, 0, 1);
    T(gWl[1], gWlt[1], LL, LL, 0, 0, 1);
    T(gWr[1], gWrt[1], LL, LL, 0, 0, 1);
    T(gWl[2], gWlt[2], LL, LL, 0, 0, 1);
    T(gWr[2], gWrt[2], LL, LL, 0, 0, 1);
    T(Wf, Wft, LL, DD, 0, 0, 1);

    // ---- factored pair-MLP projections (WMMA GEMMs) ----
    k_gemm<4, DD><<<dim3(LL / 16, (BB * NN) / 64, 1), 32, 0, stream>>>(
        plh, W01tT, b01, U, BB * NN, LL, 0, 0, 0, 0, nullptr);
    k_gemm<4, DD><<<dim3(LL / 16, (BB * NN) / 64, 1), 32, 0, stream>>>(
        plh, W01tB, nullptr, V, BB * NN, LL, 0, 0, 0, 0, nullptr);
    k_gemm<4, DD><<<dim3(LL / 16, NN / 64, BB), 32, 0, stream>>>(
        plh, We1tT, be1, Ue, NN, LL,
        (long)NN * DD, (long)LL * DD, (long)LL, (long)NN * LL, idx);
    k_gemm<4, DD><<<dim3(LL / 16, NN / 64, BB), 32, 0, stream>>>(
        plh, We1tB, nullptr, Ve, NN, LL,
        (long)NN * DD, (long)LL * DD, 0, (long)NN * LL, idx);

    // ---- pairwise adjacency sampling + self-loop mean ----
    k_pair<<<(BB * NN * NN) / 8, 256, 0, stream>>>(
        U, V, Ue, Ve, W02, b02, We2, be2, idx, mask, g_adj, adj);
    k_mean<<<1, 256, 0, stream>>>(adj, meanv);

    // ---- GATv2 stack ----
    const half_t* xh = nodesh;
    half_t* houts[3] = {hA, hB, hA};
    for (int l = 0; l < 3; ++l) {
        if (l == 0) {
            k_gemm<5, DD><<<dim3(LL / 16, NP / 80, BB), 32, 0, stream>>>(
                xh, gWlt[0], gbl[0], glb, NP, LL,
                (long)NP * DD, 0, 0, (long)NP * LL, nullptr);
            k_gemm<5, DD><<<dim3(LL / 16, NP / 80, BB), 32, 0, stream>>>(
                xh, gWrt[0], gbr[0], grb, NP, LL,
                (long)NP * DD, 0, 0, (long)NP * LL, nullptr);
        } else {
            k_gemm<5, LL><<<dim3(LL / 16, NP / 80, BB), 32, 0, stream>>>(
                xh, gWlt[l], gbl[l], glb, NP, LL,
                (long)NP * LL, 0, 0, (long)NP * LL, nullptr);
            k_gemm<5, LL><<<dim3(LL / 16, NP / 80, BB), 32, 0, stream>>>(
                xh, gWrt[l], gbr[l], grb, NP, LL,
                (long)NP * LL, 0, 0, (long)NP * LL, nullptr);
        }
        k_gatt<<<dim3(NP, BB), 256, 0, stream>>>(
            glb, grb, adj, meanv, gWe[l], gatt[l], gb[l], houts[l]);
        xh = houts[l];
    }

    // ---- final projection + softmax ----
    k_gemm<5, LL><<<dim3(DD / 16, NP / 80, BB), 32, 0, stream>>>(
        hA, Wft, bf, logits, NP, DD,
        (long)NP * LL, 0, 0, (long)NP * DD, nullptr);
    k_out<<<dim3(NN, BB), 256, 0, stream>>>(logits, (float*)d_out);
}
